// Loss_fn2_39719857553718
// MI455X (gfx1250) — matrix operand order repaired and checked by reading, not compiled, not run on hardware
//
#include <hip/hip_runtime.h>
#include <hip/hip_bf16.h>

#define D 512
#define NROWS 32768
#define BK 32
#define BM 128                    // A-side (i) block tile
#define BN 64                     // B-side (j) block tile
#define KSPLIT 16
#define KCHUNK (NROWS / KSPLIT)   // 2048
#define LDK 40                    // padded K stride in LDS (ushorts); 80B -> 16B-aligned runs

typedef __attribute__((ext_vector_type(16))) __bf16 v16bf;
typedef __attribute__((ext_vector_type(8)))  float  v8f;

// ---------------- helpers ----------------

__device__ inline v8f wmma_bf16(v16bf a, v16bf b, v8f c) {
    return __builtin_amdgcn_wmma_f32_16x16x32_bf16(
        /*neg_a=*/false, a, /*neg_b=*/false, b,
        /*c_mod=*/(short)0, c, /*reuse_a=*/false, /*reuse_b=*/false);
}

// A-operand fragment (16x32, M x K); LDS tile stored [col (M)][k].
// ISA 7.12.2 16-bit A layout:
// lanes 0-15: M=lane,    v0..3 -> K=0..7,  v4..7 -> K=16..23
// lanes16-31: M=lane-16, v0..3 -> K=8..15, v4..7 -> K=24..31
// Each 16B run is aligned (80*c + 16*i) -> 2x ds_load_b128 per fragment.
__device__ inline v16bf load_fragA(const unsigned short* lds, int col0, int lane) {
    union { v16bf h; uint4 q[2]; } r;
    const int c = col0 + (lane & 15);
    const uint4* p = (const uint4*)(lds + c * LDK);
    const int half = (lane >> 4) & 1;
    r.q[0] = p[half ? 1 : 0];   // K 0..7   | 8..15
    r.q[1] = p[half ? 3 : 2];   // K 16..23 | 24..31
    return r.h;
}

// B-operand fragment (32x16, K x N); LDS tile stored [col (N)][k].
// lanes 0-15 hold K=0..7,16..23; lanes 16-31 hold K=8..15,24..31.
__device__ inline v16bf load_fragB(const unsigned short* lds, int col0, int lane) {
    union { v16bf h; uint4 q[2]; } r;
    const int c = col0 + (lane & 15);
    const uint4* p = (const uint4*)(lds + c * LDK);
    const int half = (lane >> 4) & 1;
    r.q[0] = p[half ? 1 : 0];   // K 0..7   | 8..15
    r.q[1] = p[half ? 3 : 2];   // K 16..23 | 24..31
    return r.h;
}

// ---------------- kernel 1: softmax of weight ----------------

__global__ void lf2_softmax(const float* __restrict__ weight, float* __restrict__ w) {
    __shared__ float red[1024];
    const int tid = threadIdx.x;
    float mx = -INFINITY;
    for (int i = tid; i < NROWS; i += 1024) mx = fmaxf(mx, weight[i]);
    red[tid] = mx; __syncthreads();
    for (int s = 512; s > 0; s >>= 1) {
        if (tid < s) red[tid] = fmaxf(red[tid], red[tid + s]);
        __syncthreads();
    }
    mx = red[0]; __syncthreads();
    float sum = 0.f;
    for (int i = tid; i < NROWS; i += 1024) sum += __expf(weight[i] - mx);
    red[tid] = sum; __syncthreads();
    for (int s = 512; s > 0; s >>= 1) {
        if (tid < s) red[tid] += red[tid + s];
        __syncthreads();
    }
    const float inv = 1.0f / red[0];
    for (int i = tid; i < NROWS; i += 1024) w[i] = __expf(weight[i] - mx) * inv;
}

// ---------------- kernel 2: m[t][d] = sum_n w[n]*M_t[n,d] ----------------

__global__ void lf2_colmean(const float* __restrict__ X, const float* __restrict__ XA,
                            const float* __restrict__ XB, const float* __restrict__ w,
                            float* __restrict__ m) {
    const int b = blockIdx.x;
    const int t = b % 3;
    const int chunk = b / 3;                  // 64 chunks of 512 rows
    const float* M = (t == 0) ? X : ((t == 1) ? XA : XB);
    const int d = threadIdx.x;                // 512 threads = D
    const int n0 = chunk * 512;
    float acc = 0.f;
#pragma unroll 4
    for (int r = 0; r < 512; ++r) {
        const int n = n0 + r;
        acc += w[n] * M[(size_t)n * D + d];
    }
    atomicAdd(&m[t * D + d], acc);
}

// ---------------- kernel 3: three weighted Gram matrices via WMMA ----------------
// C0 = (w.X)^T XA ; C1 = (w.XB)^T XA ; C2 = (w.X)^T XB
// Block: 256 threads (8 waves, 4x2), tile 128x64, bf16 hi/lo split (3-term emulation).

__global__ __launch_bounds__(256)
void lf2_gram_wmma(const float* __restrict__ X, const float* __restrict__ XA,
                   const float* __restrict__ XB, const float* __restrict__ w,
                   float* __restrict__ Cbuf) {
    // planes: [0]=hi, [1]=lo ; tiles packed: Xw(128c) | XBw(128c) | XA(64c) | XB(64c)
    __shared__ unsigned short ldsbuf[2][(2 * BM + 2 * BN) * LDK];

    const int tid = threadIdx.x;
    const int b = blockIdx.x;
    const int kc = b >> 5;                  // 0..15
    const int tile = b & 31;                // 32 tiles: 4 (i) x 8 (j)
    const int i0 = (tile >> 3) << 7;        // A-side column base (0,128,256,384)
    const int j0 = (tile & 7) << 6;         // B-side column base
    const int kbase = kc * KCHUNK;

    const int lane = tid & 31;
    const int wid = tid >> 5;               // 0..7
    const int wm = (wid >> 1) << 5;         // 0,32,64,96
    const int wn = (wid & 1) << 5;          // 0,32

    const float* srcs[4] = { X, XB, XA, XB };
    const int toff[4] = { 0, BM * LDK, 2 * BM * LDK, (2 * BM + BN) * LDK };

    v8f acc[3][2][2] = {};

    const int iters = KCHUNK / BK;          // 64
    for (int it = 0; it < iters; ++it) {
        const int k0 = kbase + it * BK;

        // prefetch next K-chunk (global_prefetch_b8)
        if (it + 1 < iters && tid < 4) {
            const int cb = (tid < 2) ? i0 : j0;
            __builtin_prefetch(srcs[tid] + (size_t)(k0 + BK) * D + cb, 0, 0);
        }

        // ---- stage tiles: fp32 -> bf16 hi/lo, transposed [col][k] ----
        // unit = 4 K-rows x 4 columns -> packed ds_store_b64 per column/plane
#pragma unroll
        for (int t = 0; t < 4; ++t) {
            const float* src = srcs[t];
            const int cb = (t < 2) ? i0 : j0;
            const bool weighted = (t < 2);
            const int csh = (t < 2) ? 5 : 4;             // log2(cols/4)
            const int cmask = (1 << csh) - 1;
            const int nunit = 8 << csh;                  // (BK/4) row-quads x col4 groups
            for (int e = tid; e < nunit; e += 256) {
                const int rq = e >> csh;                 // 0..7 (row quad)
                const int c4 = (e & cmask) << 2;         // column base
                const int ka = k0 + 4 * rq;
                union { float4 v4; float f[4]; } vr[4];
                float s[4];
#pragma unroll
                for (int r = 0; r < 4; ++r) {
                    vr[r].v4 = *(const float4*)(src + (size_t)(ka + r) * D + cb + c4);
                    s[r] = weighted ? w[ka + r] : 1.0f;
                }
#pragma unroll
                for (int q = 0; q < 4; ++q) {
                    union { __bf16 h[4]; uint2 u2; } ph, pl;
#pragma unroll
                    for (int r = 0; r < 4; ++r) {
                        const float f = vr[r].f[q] * s[r];
                        const __bf16 h = (__bf16)f;
                        ph.h[r] = h;
                        pl.h[r] = (__bf16)(f - (float)h);
                    }
                    const int idx = toff[t] + (c4 + q) * LDK + 4 * rq;  // 8B aligned
                    *(uint2*)&ldsbuf[0][idx] = ph.u2;
                    *(uint2*)&ldsbuf[1][idx] = pl.u2;
                }
            }
        }
        __syncthreads();

        // ---- gather fragments (2x ds_load_b128 each) ----
        v16bf aXh[2], aXl[2], aYh[2], aYl[2];
        v16bf bAh[2], bAl[2], bBh[2], bBl[2];
#pragma unroll
        for (int mi = 0; mi < 2; ++mi) {
            const int c = wm + mi * 16;
            aXh[mi] = load_fragA(&ldsbuf[0][toff[0]], c, lane);
            aXl[mi] = load_fragA(&ldsbuf[1][toff[0]], c, lane);
            aYh[mi] = load_fragA(&ldsbuf[0][toff[1]], c, lane);
            aYl[mi] = load_fragA(&ldsbuf[1][toff[1]], c, lane);
        }
#pragma unroll
        for (int ni = 0; ni < 2; ++ni) {
            const int c = wn + ni * 16;
            bAh[ni] = load_fragB(&ldsbuf[0][toff[2]], c, lane);
            bAl[ni] = load_fragB(&ldsbuf[1][toff[2]], c, lane);
            bBh[ni] = load_fragB(&ldsbuf[0][toff[3]], c, lane);
            bBl[ni] = load_fragB(&ldsbuf[1][toff[3]], c, lane);
        }

        // ---- 36 WMMAs/wave: 3 products x 2x2 tiles x 3 emulation terms ----
#pragma unroll
        for (int mi = 0; mi < 2; ++mi) {
#pragma unroll
            for (int ni = 0; ni < 2; ++ni) {
                acc[0][mi][ni] = wmma_bf16(aXh[mi], bAh[ni], acc[0][mi][ni]);
                acc[0][mi][ni] = wmma_bf16(aXh[mi], bAl[ni], acc[0][mi][ni]);
                acc[0][mi][ni] = wmma_bf16(aXl[mi], bAh[ni], acc[0][mi][ni]);

                acc[1][mi][ni] = wmma_bf16(aYh[mi], bAh[ni], acc[1][mi][ni]);
                acc[1][mi][ni] = wmma_bf16(aYh[mi], bAl[ni], acc[1][mi][ni]);
                acc[1][mi][ni] = wmma_bf16(aYl[mi], bAh[ni], acc[1][mi][ni]);

                acc[2][mi][ni] = wmma_bf16(aXh[mi], bBh[ni], acc[2][mi][ni]);
                acc[2][mi][ni] = wmma_bf16(aXh[mi], bBl[ni], acc[2][mi][ni]);
                acc[2][mi][ni] = wmma_bf16(aXl[mi], bBh[ni], acc[2][mi][ni]);
            }
        }
        __syncthreads();
    }

    // ---- epilogue: atomic accumulate K-split partials ----
    const int half = lane >> 4;
    const int jn = lane & 15;
#pragma unroll
    for (int p = 0; p < 3; ++p)
#pragma unroll
        for (int mi = 0; mi < 2; ++mi)
#pragma unroll
            for (int ni = 0; ni < 2; ++ni)
#pragma unroll
                for (int r = 0; r < 8; ++r) {
                    const int gi = i0 + wm + mi * 16 + r + half * 8;
                    const int gj = j0 + wn + ni * 16 + jn;
                    atomicAdd(&Cbuf[(size_t)p * D * D + (size_t)gi * D + gj],
                              acc[p][mi][ni][r]);
                }
}

// ---------------- kernel 4: MSE over the three D x D results ----------------

__global__ void lf2_loss(const float* __restrict__ Cbuf, const float* __restrict__ m,
                         float* __restrict__ lossAcc) {
    __shared__ float red[256];
    const int tid = threadIdx.x;
    const size_t idx = (size_t)blockIdx.x * 256 + tid;
    const float invN = 1.0f / (float)NROWS;
    float v = 0.f;
    if (idx < (size_t)3 * D * D) {
        const int t = (int)(idx / (D * D));
        const int rem = (int)(idx % (D * D));
        const int i = rem / D;
        int a, bb;
        if (t == 0) { a = 0; bb = 1; }        // (X, X_A)
        else if (t == 1) { a = 2; bb = 1; }   // (X_B, X_A)
        else { a = 0; bb = 2; }               // (X, X_B)
        const float vec1 = Cbuf[idx] * invN;
        const float vec2 = (m[a * D + i] * invN) * (m[bb * D + i] * invN);
        const float d = vec1 - vec2;
        v = d * d;
    }
    red[tid] = v; __syncthreads();
    for (int s = 128; s > 0; s >>= 1) {
        if (tid < s) red[tid] += red[tid + s];
        __syncthreads();
    }
    if (tid == 0) atomicAdd(lossAcc, red[0]);
}

__global__ void lf2_writeout(const float* __restrict__ lossAcc, float* __restrict__ out) {
    if (threadIdx.x == 0) {
        const float l = *lossAcc * (1.0f / ((float)D * (float)D));
        out[0] = l;   // total_loss (alpha = 1)
        out[1] = l;   // loss_vec_orthogonal
    }
}

// ---------------- launcher ----------------

extern "C" void kernel_launch(void* const* d_in, const int* in_sizes, int n_in,
                              void* d_out, int out_size, void* d_ws, size_t ws_size,
                              hipStream_t stream) {
    const float* X  = (const float*)d_in[0];
    const float* XA = (const float*)d_in[1];
    const float* XB = (const float*)d_in[2];
    const float* weight = (const float*)d_in[3];
    float* out = (float*)d_out;

    float* ws = (float*)d_ws;
    float* w_soft  = ws;                        // 32768
    float* mvec    = ws + NROWS;                // 3*512
    float* Cbuf    = mvec + 3 * D;              // 3*512*512
    float* lossAcc = Cbuf + 3 * D * D;          // 1

    // zero m, C, loss regions (w is fully overwritten by softmax)
    hipMemsetAsync(mvec, 0, (size_t)(3 * D + 3 * D * D + 1) * sizeof(float), stream);

    lf2_softmax<<<1, 1024, 0, stream>>>(weight, w_soft);

    lf2_colmean<<<3 * (NROWS / 512), 512, 0, stream>>>(X, XA, XB, w_soft, mvec);

    lf2_gram_wmma<<<(D / BM) * (D / BN) * KSPLIT, 256, 0, stream>>>(
        X, XA, XB, w_soft, Cbuf);

    lf2_loss<<<(3 * D * D + 255) / 256, 256, 0, stream>>>(Cbuf, mvec, lossAcc);

    lf2_writeout<<<1, 32, 0, stream>>>(lossAcc, out);
}
